// CNFTransform_80058190397958
// MI455X (gfx1250) — matrix operand order repaired and checked by reading, not compiled, not run on hardware
//
#include <hip/hip_runtime.h>

// CNF RK4 integrator for MI455X (gfx1250, wave32, WMMA f16 16x16x32).
//
// trace(J) = d1^T * G * d2 with G = W2 .* (W3*W1d)^T  (precomputed once),
// so each vector-field eval is 4 GEMMs:
//   z1 = x@W1d + (b1 + t*W1[32,:]) ; h1=tanh ; d1=1-h1^2
//   z2 = h1@W2 + b2               ; h2=tanh ; d2=1-h2^2
//   f  = h2@W3 + b3
//   U  = d1@G ; trace = rowsum(U .* d2)
// All GEMMs run on v_wmma_f32_16x16x32_f16 with f32 accumulate.
//
// Round-3 changes: layer-1 unrolled x2 to fill the WMMA->VALU hazard slots
// with the sibling tile's work; RK4 "tmp" state eliminated (k buffer updated
// in place and passed as both input and output of vector_field -- safe since
// the input is fully staged to LDS before the output is written).

typedef _Float16 half_t;
typedef __attribute__((ext_vector_type(16))) _Float16 v16h;
typedef __attribute__((ext_vector_type(8)))  _Float16 v8h;
typedef __attribute__((ext_vector_type(8)))  float    v8f;

#define NROWS   8192
#define DDIM    32
#define HDIM    256
#define NSTEPS  4
#define WAVES   4
#define ROWS_PER_BLOCK (WAVES * 16)

// Pre-swizzled weight fragments: each fragment = 32 lanes x 16 halves = 512 halves (1KB).
#define FRAG_HALVES 512
#define W1B_FRAG0   0     // 16 frags  : B = W1d  (32x256),  [nt]
#define W2B_FRAG0   16    // 128 frags : B = W2   (256x256), [nt*8+kt]
#define GB_FRAG0    144   // 128 frags : B = G    (256x256), [nt*8+kt]
#define W3B_FRAG0   272   // 16 frags  : B = W3   (256x32),  [nt*8+kt]
#define TOTAL_FRAGS 288

__device__ __forceinline__ v8f zero8() {
  v8f z;
#pragma unroll
  for (int i = 0; i < 8; ++i) z[i] = 0.0f;
  return z;
}

// tanh(x) = 1 - 2/(exp2(x*2*log2(e)) + 1); v_exp_f32 + v_rcp_f32, saturates
// correctly for |x| large (exp2->0 => -1, exp2->inf => rcp->0 => +1).
__device__ __forceinline__ float fast_tanh(float x) {
  const float e = __builtin_amdgcn_exp2f(x * 2.8853900817779268f);
  const float r = __builtin_amdgcn_rcpf(e + 1.0f);
  return __builtin_fmaf(-2.0f, r, 1.0f);
}

__device__ __forceinline__ v8f wmma_f16(v16h a, v16h b, v8f c) {
  // (neg_a, A, neg_b, B, c_mod, C, reuse_a, reuse_b)
  return __builtin_amdgcn_wmma_f32_16x16x32_f16(false, a, false, b, (short)0, c,
                                                false, false);
}

// B fragment (KxN, 32x16, f16): lane L holds N = nt*16 + (L&15);
// halves s=0..7 hold K = 8*(L>=16) + s, s=8..15 hold K = 16 + 8*(L>=16) + (s-8). Stored fragment-ready: 32B/lane, coalesced.
__device__ __forceinline__ v16h load_bfrag(const half_t* __restrict__ base,
                                           int frag, int lane) {
  return *(const v16h*)(base + ((size_t)frag * 32 + (size_t)lane) * 16);
}

// A fragment (MxK, 16x32, f16) from row-major LDS [16][stride] halves:
// lane L holds M = L&15; halves 0..7 -> K = k0..k0+7, halves 8..15 -> K = k0+16..k0+23,
// with k0 = kt*32 + 8*(L>=16). Two contiguous ds_load_b128 per fragment.
__device__ __forceinline__ v16h load_afrag(const half_t* buf, int stride,
                                           int kt, int lane) {
  const int row = lane & 15;
  const int k0  = kt * 32 + ((lane >> 4) << 3);
  const v8h lo = *(const v8h*)(buf + row * stride + k0);
  const v8h hi = *(const v8h*)(buf + row * stride + k0 + 16);
  v16h r;
#pragma unroll
  for (int i = 0; i < 8; ++i) { r[i] = lo[i]; r[8 + i] = hi[i]; }
  return r;
}

// ---------------------------------------------------------------------------
// Pre-swizzle: W1d/W2/W3 -> f16 B fragments; also build G = W2 .* (W3*W1d)^T.
// One block per fragment, one thread per half (512 threads).
// ---------------------------------------------------------------------------
__global__ void cnf_swizzle_weights(const float* __restrict__ W1,   // (33,256)
                                    const float* __restrict__ W2,   // (256,256)
                                    const float* __restrict__ W3,   // (256,32)
                                    half_t* __restrict__ ws) {
  const int bid  = blockIdx.x;
  const int tid  = threadIdx.x;      // 0..511
  const int lane = tid >> 4;         // 0..31
  const int slot = tid & 15;         // 0..15
  const int kk   = (slot & 7) + ((lane >= 16) ? 8 : 0) + ((slot >> 3) << 4);  // K within 32-wide K-tile
  const int nn   = lane & 15;                        // N within 16-wide N-tile
  float val;
  if (bid < W2B_FRAG0) {                       // W1d fragments
    const int nt = bid;
    val = W1[kk * HDIM + nt * 16 + nn];        // rows 0..31 of W1 (drop t-row)
  } else if (bid < GB_FRAG0) {                 // W2 fragments
    const int f = bid - W2B_FRAG0, nt = f >> 3, kt = f & 7;
    const int a = kt * 32 + kk, b = nt * 16 + nn;
    val = W2[a * HDIM + b];
  } else if (bid < W3B_FRAG0) {                // G fragments
    const int f = bid - GB_FRAG0, nt = f >> 3, kt = f & 7;
    const int a = kt * 32 + kk, b = nt * 16 + nn;
    float s = 0.0f;
#pragma unroll 8
    for (int j = 0; j < DDIM; ++j) s += W3[b * DDIM + j] * W1[j * HDIM + a];
    val = W2[a * HDIM + b] * s;
  } else {                                     // W3 fragments
    const int f = bid - W3B_FRAG0, nt = f >> 3, kt = f & 7;
    const int K = kt * 32 + kk, n = nt * 16 + nn;
    val = W3[K * DDIM + n];
  }
  ws[(size_t)bid * FRAG_HALVES + (size_t)lane * 16 + slot] = (half_t)val;
}

// ---------------------------------------------------------------------------
// One vector-field evaluation for a 16-row tile (one wave).
// in/out use C-fragment layout: vgpr r, lanes 0-15 -> (M=r, N=lane),
// lanes 16-31 -> (M=8+r, N=lane-16).
// NOTE: `in` and `fout` may alias: `in` is fully consumed (staged to LDS) at
// entry, `fout` is only written at the end.
// ---------------------------------------------------------------------------
__device__ __forceinline__ void vector_field(
    float t, const float (&in)[2][8], float (&fout)[2][8], float (&trout)[8],
    const half_t* __restrict__ W1B, const half_t* __restrict__ W2B,
    const half_t* __restrict__ GB,  const half_t* __restrict__ W3B,
    const float* __restrict__ b1, const float* __restrict__ b2,
    const float* __restrict__ b3, const float* __restrict__ w1t,
    half_t* buf, int lane) {
  const int hi8 = (lane >> 4) << 3;   // 0 or 8: row offset of this lane's C values
  const int nn  = lane & 15;

  // ---- stage x tile (16x32 f16, row-major stride 32) -----------------------
#pragma unroll
  for (int nt = 0; nt < 2; ++nt)
#pragma unroll
    for (int r = 0; r < 8; ++r)
      buf[(hi8 + r) * DDIM + nt * 16 + nn] = (half_t)in[nt][r];
  const v16h xA = load_afrag(buf, DDIM, 0, lane);   // DS is in-order per wave

  // ---- layer 1: z1 = x@W1d + b1 + t*W1[32,:]; h1 = tanh --------------------
  // Unroll x2: the sibling tile's VALU tail fills the WMMA->VALU hazard slots.
#pragma unroll 2
  for (int nt = 0; nt < 16; ++nt) {
    v8f acc = zero8();
    acc = wmma_f16(xA, load_bfrag(W1B, nt, lane), acc);
    const int col = nt * 16 + nn;
    const float bias = b1[col] + t * w1t[col];   // L0/L2-resident after eval 1
#pragma unroll
    for (int r = 0; r < 8; ++r) {
      const float h = fast_tanh(acc[r] + bias);
      buf[(hi8 + r) * HDIM + col] = (half_t)h;   // stage h1
    }
  }
  // h1 A-fragments; derive d1 = 1 - h1^2 in-register (same A layout).
  v16h h1A[8], d1A[8];
  v16h one16;
#pragma unroll
  for (int i = 0; i < 16; ++i) one16[i] = (_Float16)1.0f;
#pragma unroll
  for (int kt = 0; kt < 8; ++kt) {
    h1A[kt] = load_afrag(buf, HDIM, kt, lane);
    d1A[kt] = one16 - h1A[kt] * h1A[kt];         // v_pk f16 ops
  }

  // ---- layer 2 (h1@W2) fused with trace GEMM (d1@G) ------------------------
#pragma unroll
  for (int r = 0; r < 8; ++r) trout[r] = 0.0f;
#pragma unroll 1   // rolled: two independent 8-deep WMMA chains interleave
  for (int nt = 0; nt < 16; ++nt) {
    v8f acc = zero8();
#pragma unroll
    for (int kt = 0; kt < 8; ++kt)
      acc = wmma_f16(h1A[kt], load_bfrag(W2B, nt * 8 + kt, lane), acc);
    v8f u = zero8();
#pragma unroll
    for (int kt = 0; kt < 8; ++kt)
      u = wmma_f16(d1A[kt], load_bfrag(GB, nt * 8 + kt, lane), u);
    const int col = nt * 16 + nn;
    const float bias = b2[col];
#pragma unroll
    for (int r = 0; r < 8; ++r) {
      const float h = fast_tanh(acc[r] + bias);
      const float d = __builtin_fmaf(-h, h, 1.0f);   // d2, C layout
      trout[r] = __builtin_fmaf(u[r], d, trout[r]);  // U .* d2 (matching M,N)
      buf[(hi8 + r) * HDIM + col] = (half_t)h;       // stage h2
    }
  }
  // h2 A-fragments (h1A/d1A dead here; allocator reuses their registers).
  v16h h2A[8];
#pragma unroll
  for (int kt = 0; kt < 8; ++kt) h2A[kt] = load_afrag(buf, HDIM, kt, lane);

  // ---- layer 3: f = h2@W3 + b3 ---------------------------------------------
#pragma unroll
  for (int nt = 0; nt < 2; ++nt) {
    v8f acc = zero8();
#pragma unroll
    for (int kt = 0; kt < 8; ++kt)
      acc = wmma_f16(h2A[kt], load_bfrag(W3B, nt * 8 + kt, lane), acc);
    const float bias = b3[nt * 16 + nn];
#pragma unroll
    for (int r = 0; r < 8; ++r) fout[nt][r] = acc[r] + bias;
  }

  // ---- reduce trace over the 16-lane N-group (xor shuffles stay in-half) ---
#pragma unroll
  for (int r = 0; r < 8; ++r) {
    float v = trout[r];
    v += __shfl_xor(v, 1, 32);
    v += __shfl_xor(v, 2, 32);
    v += __shfl_xor(v, 4, 32);
    v += __shfl_xor(v, 8, 32);
    trout[r] = v;   // full trace for row (r + 8*(lane>=16)), replicated
  }
}

// ---------------------------------------------------------------------------
// Persistent RK4 integrator: one 16-row tile per wave, state lives in VGPRs.
// ---------------------------------------------------------------------------
__global__ __launch_bounds__(WAVES * 32)
void cnf_integrate(const float* __restrict__ x,
                   const float* __restrict__ W1,
                   const float* __restrict__ b1,
                   const float* __restrict__ b2,
                   const float* __restrict__ b3,
                   const half_t* __restrict__ ws,
                   float* __restrict__ outS,
                   float* __restrict__ outL) {
  __shared__ half_t lds[WAVES][16 * HDIM];   // 8KB per wave, private staging
  const int lane = threadIdx.x & 31;
  const int wv   = threadIdx.x >> 5;
  half_t* buf    = &lds[wv][0];
  const int row0 = (blockIdx.x * WAVES + wv) * 16;
  const int hi   = lane >> 4;
  const int nn   = lane & 15;

  const half_t* W1B = ws + (size_t)W1B_FRAG0 * FRAG_HALVES;
  const half_t* W2B = ws + (size_t)W2B_FRAG0 * FRAG_HALVES;
  const half_t* GB  = ws + (size_t)GB_FRAG0  * FRAG_HALVES;
  const half_t* W3B = ws + (size_t)W3B_FRAG0 * FRAG_HALVES;
  const float*  w1t = W1 + (size_t)DDIM * HDIM;   // t-row of W1

  // Load state tile (C layout) + logdet.
  float sC[2][8];
#pragma unroll
  for (int nt = 0; nt < 2; ++nt)
#pragma unroll
    for (int r = 0; r < 8; ++r)
      sC[nt][r] = x[(size_t)(row0 + r + 8 * hi) * DDIM + nt * 16 + nn];
  float ld8[8];
#pragma unroll
  for (int r = 0; r < 8; ++r) ld8[r] = 0.0f;

  const float dt = 1.0f / NSTEPS;
  float kC[2][8], kT[8], aC[2][8], aT[8];

  for (int step = 0; step < NSTEPS; ++step) {
    const float t0 = step * dt;
    // k1
    vector_field(t0, sC, kC, kT, W1B, W2B, GB, W3B, b1, b2, b3, w1t, buf, lane);
#pragma unroll
    for (int nt = 0; nt < 2; ++nt)
#pragma unroll
      for (int r = 0; r < 8; ++r) {
        aC[nt][r] = kC[nt][r];
        kC[nt][r] = sC[nt][r] + 0.5f * dt * kC[nt][r];   // eval point for k2
      }
#pragma unroll
    for (int r = 0; r < 8; ++r) aT[r] = kT[r];
    // k2 (in-place: kC is both eval point and result)
    vector_field(t0 + 0.5f * dt, kC, kC, kT, W1B, W2B, GB, W3B, b1, b2, b3, w1t, buf, lane);
#pragma unroll
    for (int nt = 0; nt < 2; ++nt)
#pragma unroll
      for (int r = 0; r < 8; ++r) {
        aC[nt][r] += 2.0f * kC[nt][r];
        kC[nt][r] = sC[nt][r] + 0.5f * dt * kC[nt][r];   // eval point for k3
      }
#pragma unroll
    for (int r = 0; r < 8; ++r) aT[r] += 2.0f * kT[r];
    // k3
    vector_field(t0 + 0.5f * dt, kC, kC, kT, W1B, W2B, GB, W3B, b1, b2, b3, w1t, buf, lane);
#pragma unroll
    for (int nt = 0; nt < 2; ++nt)
#pragma unroll
      for (int r = 0; r < 8; ++r) {
        aC[nt][r] += 2.0f * kC[nt][r];
        kC[nt][r] = sC[nt][r] + dt * kC[nt][r];          // eval point for k4
      }
#pragma unroll
    for (int r = 0; r < 8; ++r) aT[r] += 2.0f * kT[r];
    // k4
    vector_field(t0 + dt, kC, kC, kT, W1B, W2B, GB, W3B, b1, b2, b3, w1t, buf, lane);
#pragma unroll
    for (int nt = 0; nt < 2; ++nt)
#pragma unroll
      for (int r = 0; r < 8; ++r)
        sC[nt][r] += (dt / 6.0f) * (aC[nt][r] + kC[nt][r]);
#pragma unroll
    for (int r = 0; r < 8; ++r) ld8[r] -= (dt / 6.0f) * (aT[r] + kT[r]);
  }

  // Write outputs.
#pragma unroll
  for (int nt = 0; nt < 2; ++nt)
#pragma unroll
    for (int r = 0; r < 8; ++r)
      outS[(size_t)(row0 + r + 8 * hi) * DDIM + nt * 16 + nn] = sC[nt][r];
#pragma unroll
  for (int r = 0; r < 8; ++r)
    if (nn == r) outL[row0 + r + 8 * hi] = ld8[r];
}

// ---------------------------------------------------------------------------
extern "C" void kernel_launch(void* const* d_in, const int* in_sizes, int n_in,
                              void* d_out, int out_size, void* d_ws,
                              size_t ws_size, hipStream_t stream) {
  const float* x  = (const float*)d_in[0];
  const float* W1 = (const float*)d_in[1];
  const float* b1 = (const float*)d_in[2];
  const float* W2 = (const float*)d_in[3];
  const float* b2 = (const float*)d_in[4];
  const float* W3 = (const float*)d_in[5];
  const float* b3 = (const float*)d_in[6];
  half_t* ws = (half_t*)d_ws;   // needs TOTAL_FRAGS*512*2 = 288KB
  float* outS = (float*)d_out;
  float* outL = outS + (size_t)NROWS * DDIM;

  cnf_swizzle_weights<<<TOTAL_FRAGS, 512, 0, stream>>>(W1, W2, W3, ws);
  cnf_integrate<<<NROWS / ROWS_PER_BLOCK, WAVES * 32, 0, stream>>>(
      x, W1, b1, b2, b3, ws, outS, outL);
}
